// AugmentedTensor_3324304687347
// MI455X (gfx1250) — matrix operand order repaired and checked by reading, not compiled, not run on hardware
//
#include <hip/hip_runtime.h>

typedef _Float16 half_t;
typedef __attribute__((ext_vector_type(16))) _Float16 v16h;
typedef __attribute__((ext_vector_type(8)))  _Float16 v8h;
typedef __attribute__((ext_vector_type(4)))  _Float16 v4h;
typedef __attribute__((ext_vector_type(8)))  float    v8f;
typedef __attribute__((ext_vector_type(4)))  float    v4f;
typedef __attribute__((ext_vector_type(8)))  unsigned short us8;
typedef __attribute__((ext_vector_type(4)))  unsigned short us4;

#define BSZ   4096
#define NA    4
#define DIM   128
#define DP1   129
#define OUTD  128

#define BROW  136            // halves per o-row of a main B buffer (128 + pad)
#define BUFH  (128 * BROW)   // halves per main B buffer (row pad -> 64-bank stripes)
#define TROW  168            // halves per o-row of tail buffer (160 + pad)

static __device__ __forceinline__ v16h cat16(v8h lo, v8h hi) {
  return __builtin_shufflevector(lo, hi, 0,1,2,3,4,5,6,7,8,9,10,11,12,13,14,15);
}
#define WMMA_F16(Afrag, Bfrag, Cacc) \
  __builtin_amdgcn_wmma_f32_16x16x32_f16(false, (Afrag), false, (Bfrag), (short)0, (Cacc), false, false)

template <bool USEF16>
__global__ __launch_bounds__(128, 1)
void bilinear_wmma_kernel(const float* __restrict__ x0,
                          const float* __restrict__ x1,
                          const float* __restrict__ T32,
                          const half_t* __restrict__ T16,
                          float* __restrict__ out)
{
  extern __shared__ half_t smem[];
  const int head = blockIdx.y;
  const int rb   = blockIdx.x * 128;          // 128 batch rows per WG
  const int tid  = threadIdx.x;
  const int w    = tid >> 5;                  // wave id (4 waves)
  const int ln   = tid & 31;
  const int n    = ln & 15;                   // A-row (M) / B,C col (N) in tile
  const int hk   = ln >> 4;                   // lane half (K / C-row half)

  const int row0 = rb + w*32 + n;             // wave's m-subtile 0 row
  const int row1 = row0 + 16;                 // m-subtile 1 row

  // ---- A-operand source: x1 rows rearranged to WMMA 16x32 f16 A layout ----
  v16h ax1[2][4];
  #pragma unroll
  for (int ms = 0; ms < 2; ++ms) {
    const float* xp = x1 + ((size_t)(row0 + ms*16) * NA + head) * DIM;
    #pragma unroll
    for (int t = 0; t < 8; ++t) {
      const int jb = hk*8 + 16*t;
      v4f f0 = *(const v4f*)(xp + jb);
      v4f f1 = *(const v4f*)(xp + jb + 4);
      const int vi = t >> 1, eo = (t & 1) * 8;
      #pragma unroll
      for (int e = 0; e < 4; ++e) {
        ax1[ms][vi][eo + e]     = (half_t)f0[e];
        ax1[ms][vi][eo + 4 + e] = (half_t)f1[e];
      }
    }
  }

  v8f acc[2][8];
  #pragma unroll
  for (int ms = 0; ms < 2; ++ms)
    #pragma unroll
    for (int nt = 0; nt < 8; ++nt)
      acc[ms][nt] = {};

  // staging assignment: thread owns j-quad [4jq,4jq+4) x o-range [og,og+32)
  const int jq = tid & 31;
  const int og = (tid >> 5) * 32;

  // B fragment load (k linearizes jb,nt): lane (n,hk) reads 16 j at fixed o
  auto ldFrag = [&](const half_t* B, int k) -> v16h {
    const int jb = k >> 3, nt = k & 7;
    const half_t* bp = B + (nt*16 + n)*BROW + jb*32 + hk*8;
    return cat16(*(const v8h*)bp, *(const v8h*)(bp + 16));
  };

  // staging chunk helpers (f16 source path); src = slab base for this thread
  auto ldChunkF16 = [&](const unsigned short* src, int g, us8* cq) {
    const unsigned short* p = src + 8*g;
    cq[0] = *(const us8*)(p);
    cq[1] = *(const us8*)(p + OUTD);
    cq[2] = *(const us8*)(p + 2*OUTD);
    cq[3] = *(const us8*)(p + 3*OUTD);
  };
  auto stChunk = [&](int bufs, int g, const us8* cq) {
    unsigned short* dstb = (unsigned short*)smem + bufs*BUFH;
    #pragma unroll
    for (int q = 0; q < 8; ++q) {
      us4 v; v[0] = cq[0][q]; v[1] = cq[1][q]; v[2] = cq[2][q]; v[3] = cq[3][q];
      *(us4*)(dstb + (og + 8*g + q)*BROW + jq*4) = v;   // b64, conflict-free
    }
  };

  // f32-source staging (fallback instantiation only)
  auto stageF32 = [&](int inext, int bufs) {
    #pragma unroll
    for (int oc = 0; oc < 32; oc += 8) {
      const int o0 = og + oc;
      const float* p = T32 + (((size_t)head*DP1 + inext)*DP1 + 4*jq)*OUTD + o0;
      v4f r0a = *(const v4f*)(p);           v4f r0b = *(const v4f*)(p + 4);
      v4f r1a = *(const v4f*)(p + OUTD);    v4f r1b = *(const v4f*)(p + OUTD + 4);
      v4f r2a = *(const v4f*)(p + 2*OUTD);  v4f r2b = *(const v4f*)(p + 2*OUTD + 4);
      v4f r3a = *(const v4f*)(p + 3*OUTD);  v4f r3b = *(const v4f*)(p + 3*OUTD + 4);
      #pragma unroll
      for (int q = 0; q < 8; ++q) {
        const int qq = q & 3;
        const float f0 = (q < 4) ? r0a[qq] : r0b[qq];
        const float f1 = (q < 4) ? r1a[qq] : r1b[qq];
        const float f2 = (q < 4) ? r2a[qq] : r2b[qq];
        const float f3 = (q < 4) ? r3a[qq] : r3b[qq];
        v4h v; v[0] = (half_t)f0; v[1] = (half_t)f1; v[2] = (half_t)f2; v[3] = (half_t)f3;
        *(v4h*)(smem + bufs*BUFH + (o0 + q)*BROW + jq*4) = v;
      }
    }
  };

  const float* x0p0 = x0 + ((size_t)row0 * NA + head) * DIM;
  const float* x0p1 = x0 + ((size_t)row1 * NA + head) * DIM;

  // per-thread staging source base (f16 path); advances by one slab per step
  const unsigned short* TuBase = nullptr;
  if constexpr (USEF16)
    TuBase = (const unsigned short*)T16
           + (((size_t)head*DP1)*DP1 + 4*jq)*OUTD + og;

  // prologue: stage slab 0 into buffer 0
  if constexpr (USEF16) {
    us8 cq[4];
    #pragma unroll
    for (int g = 0; g < 4; ++g) { ldChunkF16(TuBase, 0, cq); ldChunkF16(TuBase, g, cq); stChunk(0, g, cq); }
  } else {
    stageF32(0, 0);
  }
  __syncthreads();

  float fa = x0p0[0], fb = x0p1[0];   // x0 scalars prefetched one step ahead
  for (int i = 0; i < DP1; ++i) {
    const half_t s0 = (i < DIM) ? (half_t)fa : (half_t)1.0f;
    const half_t s1 = (i < DIM) ? (half_t)fb : (half_t)1.0f;
    const int ipre = (i + 1 < DIM) ? (i + 1) : (DIM - 1);
    fa = x0p0[ipre]; fb = x0p1[ipre];
    const int inext = (i + 1 < DP1) ? (i + 1) : DIM;  // last iter restages 128 (harmless)

    const half_t* B = smem + (i & 1) * BUFH;
    const int bufs = (i + 1) & 1;

    if constexpr (USEF16) {
      // fused step: WMMA stream + depth-3 fragment pipeline + interleaved staging
      const unsigned short* src = TuBase + (size_t)inext * (DP1 * OUTD);
      us8 cq[4];
      v16h pf0 = ldFrag(B, 0);
      v16h pf1 = ldFrag(B, 1);
      v16h pf2 = ldFrag(B, 2);
      v16h af0, af1;
      #pragma unroll
      for (int k = 0; k < 32; ++k) {
        if ((k & 7) == 0) {
          const int jb = k >> 3;
          af0 = ax1[0][jb] * s0;          // rank-1 A-tile: x0[m,i] * x1[m,j]
          af1 = ax1[1][jb] * s1;
          ldChunkF16(src, jb, cq);        // issue next-slab global loads early
        }
        if ((k & 7) == 6) stChunk(bufs, k >> 3, cq);  // ~6 WMMA-pairs of cover
        const int sl = k % 3;
        v16h cur = (sl == 0) ? pf0 : (sl == 1) ? pf1 : pf2;
        if (k + 3 < 32) {
          v16h nf = ldFrag(B, k + 3);
          if (sl == 0) pf0 = nf; else if (sl == 1) pf1 = nf; else pf2 = nf;
        }
        const int nt = k & 7;
        acc[0][nt] = WMMA_F16(af0, cur, acc[0][nt]);
        acc[1][nt] = WMMA_F16(af1, cur, acc[1][nt]);
      }
    } else {
      // compute-only, then stage (fallback path; separate instantiation)
      v16h pf0 = ldFrag(B, 0);
      v16h pf1 = ldFrag(B, 1);
      v16h pf2 = ldFrag(B, 2);
      v16h af0, af1;
      #pragma unroll
      for (int k = 0; k < 32; ++k) {
        if ((k & 7) == 0) {
          const int jb = k >> 3;
          af0 = ax1[0][jb] * s0;
          af1 = ax1[1][jb] * s1;
        }
        const int sl = k % 3;
        v16h cur = (sl == 0) ? pf0 : (sl == 1) ? pf1 : pf2;
        if (k + 3 < 32) {
          v16h nf = ldFrag(B, k + 3);
          if (sl == 0) pf0 = nf; else if (sl == 1) pf1 = nf; else pf2 = nf;
        }
        const int nt = k & 7;
        acc[0][nt] = WMMA_F16(af0, cur, acc[0][nt]);
        acc[1][nt] = WMMA_F16(af1, cur, acc[1][nt]);
      }
      stageF32(inext, bufs);
    }
    __syncthreads();
  }

  // ---- tail: j = 128 column -> GEMM over k=i (129 valid, padded to 160) ----
  {
    unsigned short* rowp = (unsigned short*)smem + tid * TROW;  // row o = tid
    us8 z;
    #pragma unroll
    for (int e = 0; e < 8; ++e) z[e] = 0;
    #pragma unroll
    for (int q = 0; q < TROW; q += 8) *(us8*)(rowp + q) = z;
  }
  __syncthreads();
  if (tid <= 64) {                       // gather B2[k,o] = T[a,k,128,o]
    const int k0 = 2*tid, k1 = k0 + 1;
    const bool v1 = (k1 <= DIM);
    const float* pk0 = T32 + (((size_t)head*DP1 + k0)*DP1 + DIM)*OUTD;
    const float* pk1 = T32 + (((size_t)head*DP1 + (v1 ? k1 : k0))*DP1 + DIM)*OUTD;
    for (int o = 0; o < OUTD; ++o) {
      smem[o*TROW + k0] = (half_t)pk0[o];
      smem[o*TROW + k1] = v1 ? (half_t)pk1[o] : (half_t)0.0f;
    }
  }
  __syncthreads();

  const float* xps[2] = { x0p0, x0p1 };
  #pragma unroll
  for (int ms = 0; ms < 2; ++ms) {
    const float* xp = xps[ms];
    v16h ab[5];
    #pragma unroll
    for (int kb = 0; kb < 4; ++kb) {     // A = x0 rearranged along k=i
      v16h v;
      #pragma unroll
      for (int e = 0; e < 16; ++e) {
        const int i = kb*32 + ((e < 8) ? (hk*8 + e) : (16 + hk*8 + (e - 8)));
        v[e] = (half_t)xp[i];
      }
      ab[kb] = v;
    }
    {                                    // ones block: only k=128 element = 1
      v16h v;
      #pragma unroll
      for (int e = 0; e < 16; ++e) v[e] = (half_t)0.0f;
      if (hk == 0) v[0] = (half_t)1.0f;
      ab[4] = v;
    }
    #pragma unroll
    for (int kb = 0; kb < 5; ++kb) {
      #pragma unroll
      for (int nt = 0; nt < 8; ++nt) {
        const half_t* bp = smem + (nt*16 + n)*TROW + kb*32 + hk*8;
        v16h bf = cat16(*(const v8h*)bp, *(const v8h*)(bp + 16));
        acc[ms][nt] = WMMA_F16(ab[kb], bf, acc[ms][nt]);
      }
    }
  }

  // ---- store C: lane holds C[hk*8 + r, n] in acc[..][nt][r] ----
  #pragma unroll
  for (int ms = 0; ms < 2; ++ms) {
    const int rbase = rb + w*32 + ms*16 + hk*8;
    #pragma unroll
    for (int nt = 0; nt < 8; ++nt) {
      float* po = out + ((size_t)rbase * NA + head) * OUTD + nt*16 + n;
      #pragma unroll
      for (int r = 0; r < 8; ++r)
        po[(size_t)r * NA * OUTD] = acc[ms][nt][r];
    }
  }
}

// One-shot T f32 -> f16 conversion into workspace (T then lives in 192MB L2).
__global__ void cvt_f32_f16(const float* __restrict__ src,
                            half_t* __restrict__ dst, int n8) {
  const int idx = blockIdx.x * blockDim.x + threadIdx.x;
  if (idx >= n8) return;
  const v4f* s = (const v4f*)src + (size_t)idx * 2;
  v4f f0 = s[0], f1 = s[1];
  v8h h;
  #pragma unroll
  for (int e = 0; e < 4; ++e) { h[e] = (half_t)f0[e]; h[e + 4] = (half_t)f1[e]; }
  *(v8h*)(dst + (size_t)idx * 8) = h;
}

extern "C" void kernel_launch(void* const* d_in, const int* in_sizes, int n_in,
                              void* d_out, int out_size, void* d_ws, size_t ws_size,
                              hipStream_t stream) {
  (void)in_sizes; (void)n_in; (void)out_size;
  const float* x0 = (const float*)d_in[0];
  const float* x1 = (const float*)d_in[1];
  const float* T  = (const float*)d_in[2];
  float* out = (float*)d_out;

  const size_t tElems = (size_t)NA * DP1 * DP1 * OUTD;   // 8,520,192
  const size_t tBytes = tElems * sizeof(half_t);         // ~16.3 MB
  const bool useF16 = (ws_size >= tBytes);

  const int ldsBytes = (int)(2 * BUFH * sizeof(half_t)); // 69,632 B (< 320KB/WGP)
  dim3 grid(BSZ / 128, NA);

  if (useF16) {
    half_t* T16 = (half_t*)d_ws;
    const int n8 = (int)(tElems / 8);
    cvt_f32_f16<<<(n8 + 255) / 256, 256, 0, stream>>>(T, T16, n8);
    (void)hipFuncSetAttribute(
        reinterpret_cast<const void*>(&bilinear_wmma_kernel<true>),
        hipFuncAttributeMaxDynamicSharedMemorySize, ldsBytes);
    bilinear_wmma_kernel<true><<<grid, 128, ldsBytes, stream>>>(x0, x1, T, T16, out);
  } else {
    (void)hipFuncSetAttribute(
        reinterpret_cast<const void*>(&bilinear_wmma_kernel<false>),
        hipFuncAttributeMaxDynamicSharedMemorySize, ldsBytes);
    bilinear_wmma_kernel<false><<<grid, 128, ldsBytes, stream>>>(x0, x1, T, nullptr, out);
  }
}
